// MambaBlock_48077863911676
// MI455X (gfx1250) — matrix operand order repaired and checked by reading, not compiled, not run on hardware
//
#include <hip/hip_runtime.h>
#include <hip/hip_bf16.h>
#include <math.h>

// ---------------------------------------------------------------------------
// Mamba block for MI455X (gfx1250, wave32, WMMA + Tensor Data Mover).
// Pipeline: cvt(+weight transpose)->GEMM1->conv+SiLU->GEMM2->pad->GEMM3->
//           scan->gate->GEMM4.  All GEMMs: bf16 WMMA, f32 accumulate,
//           TDM-staged double-buffered LDS tiles.
// ---------------------------------------------------------------------------

typedef __attribute__((ext_vector_type(16))) __bf16   v16bf;
typedef __attribute__((ext_vector_type(8)))  float    v8f;
typedef __attribute__((ext_vector_type(4)))  uint32_t u32x4;
typedef __attribute__((ext_vector_type(8)))  uint32_t u32x8;

static_assert(sizeof(__bf16) == 2, "bf16 must be 2 bytes");

#define D_MODEL 768
#define D_INNER 1536
#define DT_RANK 48
#define D_STATE 16
#define D_CONV  4
#define B_SZ    2
#define L_SEQ   4096
#define M_ROWS  (B_SZ * L_SEQ)      // 8192 tokens
#define N_XZ    (2 * D_INNER)       // 3072
#define N_XDBL_PAD 128              // 80 -> 128
#define K_DT_PAD   64               // 48 -> 64

// ---------------------------------------------------------------------------
// Tensor Data Mover: 2D tile load Global -> LDS (ISA ch.8 descriptor, 2-group
// form via inline asm; portable across ROCm7.2 / amdgpu-toolchain builtins).
// Tile: tile_d1 rows x tile_d0 bf16 elements, source row pitch = stride
// elements, packed contiguously into LDS at lds_off.
// ---------------------------------------------------------------------------
__device__ __forceinline__ void tdm_load_tile_bf16(uint32_t lds_off,
                                                   const __bf16* gsrc,
                                                   uint32_t tile_d0,
                                                   uint32_t tile_d1,
                                                   uint32_t stride) {
  uint64_t ga = (uint64_t)(uintptr_t)gsrc;
  u32x4 g0;
  g0[0] = 1u;                                             // count=1, user D#
  g0[1] = lds_off;                                        // lds_addr[31:0]
  g0[2] = (uint32_t)ga;                                   // global_addr[31:0]
  g0[3] = (uint32_t)((ga >> 32) & 0x01FFFFFFu) | (2u << 30); // addr[56:32]|type=2
  u32x8 g1;
  g1[0] = (1u << 16);                                     // wg_mask=0, data_size=1(2B)
  g1[1] = (tile_d0 & 0xFFFFu) << 16;                      // tensor_dim0[15:0]
  g1[2] = (tile_d0 >> 16) | ((tile_d1 & 0xFFFFu) << 16);  // td0[31:16]|td1[15:0]
  g1[3] = (tile_d1 >> 16) | (tile_d0 << 16);              // td1[31:16]|tile_dim0
  g1[4] = tile_d1;                                        // tile_dim1 | tile_dim2=0
  g1[5] = stride;                                         // tensor_dim0_stride[31:0]
  g1[6] = 0u;                                             // stride[47:32]|dim1_stride lo
  g1[7] = 0u;
  asm volatile("tensor_load_to_lds %0, %1" :: "s"(g0), "s"(g1) : "memory");
}

// ---------------------------------------------------------------------------
// Tiled bf16 WMMA GEMM: C[MxN](f32) = A[MxK](bf16,row-major) * Bt[NxK](bf16)
// (B pre-transposed: K contiguous).  M%128==0, N%128==0, K%32==0.
// 256 threads = 8 wave32 waves (4x2); wave tile 32x64 = 2x4 16x16x32 WMMAs.
// TDM-staged double-buffered LDS tiles, 64B pitch -> all frag reads b128.
// ---------------------------------------------------------------------------
__global__ __launch_bounds__(256)
void gemm_bf16_wmma(const __bf16* __restrict__ A, const __bf16* __restrict__ Bt,
                    float* __restrict__ C, int M, int N, int K) {
  __shared__ __bf16 As[2][128][32];
  __shared__ __bf16 Bs[2][128][32];

  const int t     = threadIdx.x;
  const int lane  = t & 31;        // wave32
  const int w     = t >> 5;        // 8 waves
  const int waveM = w >> 1;        // 0..3
  const int waveN = w & 1;         // 0..1
  const int lm    = lane & 15;
  const int lh    = lane >> 4;     // lane half (0/1)
  const int m0    = blockIdx.y * 128;
  const int n0    = blockIdx.x * 128;
  const bool issuer = (w == 0);    // wave 0 drives the TDM

  union F16 { uint4 v[2]; v16bf f; };

  v8f acc[2][4];
  const v8f vzero = {0.f, 0.f, 0.f, 0.f, 0.f, 0.f, 0.f, 0.f};
#pragma unroll
  for (int mi = 0; mi < 2; ++mi)
#pragma unroll
    for (int ni = 0; ni < 4; ++ni) acc[mi][ni] = vzero;

  const int nsteps = K >> 5;
  if (issuer) {
    tdm_load_tile_bf16((uint32_t)(uintptr_t)&As[0][0][0], A + (size_t)m0 * K,
                       32u, 128u, (uint32_t)K);
    tdm_load_tile_bf16((uint32_t)(uintptr_t)&Bs[0][0][0], Bt + (size_t)n0 * K,
                       32u, 128u, (uint32_t)K);
  }

  int buf = 0;
  for (int s = 0; s < nsteps; ++s) {
    if (issuer) __builtin_amdgcn_s_wait_tensorcnt(0);
    __syncthreads();                               // tile[buf] ready for all
    if (issuer && (s + 1) < nsteps) {              // prefetch next k-tile
      int k1 = (s + 1) << 5;
      tdm_load_tile_bf16((uint32_t)(uintptr_t)&As[buf ^ 1][0][0],
                         A + (size_t)m0 * K + k1, 32u, 128u, (uint32_t)K);
      tdm_load_tile_bf16((uint32_t)(uintptr_t)&Bs[buf ^ 1][0][0],
                         Bt + (size_t)n0 * K + k1, 32u, 128u, (uint32_t)K);
    }

    // A fragments: elem e -> k = lh*8 + (e&7) + (e>=8)*16
    F16 af[2];
#pragma unroll
    for (int mi = 0; mi < 2; ++mi) {
      const __bf16* p = &As[buf][waveM * 32 + mi * 16 + lm][lh * 8];
      af[mi].v[0] = *(const uint4*)p;              // k = lh*8 + 0..7
      af[mi].v[1] = *(const uint4*)(p + 16);       // k = 16 + lh*8 + 0..7
    }
    // B fragments (K contiguous in Bs): elem e -> k = lh*8 + (e&7) + (e>=8)*16
#pragma unroll
    for (int ni = 0; ni < 4; ++ni) {
      F16 bf;
      const __bf16* p = &Bs[buf][waveN * 64 + ni * 16 + lm][lh * 8];
      bf.v[0] = *(const uint4*)p;
      bf.v[1] = *(const uint4*)(p + 16);
#pragma unroll
      for (int mi = 0; mi < 2; ++mi)
        acc[mi][ni] = __builtin_amdgcn_wmma_f32_16x16x32_bf16(
            false, af[mi].f, false, bf.f, (short)0, acc[mi][ni], false, false);
    }
    buf ^= 1;
    // no tail barrier needed: tile[buf_old] is only overwritten by the TDM
    // issue that follows the NEXT top barrier, which orders after this compute.
  }

  // epilogue: C/D layout -> row = v + lh*8, col = lm
#pragma unroll
  for (int mi = 0; mi < 2; ++mi)
#pragma unroll
    for (int ni = 0; ni < 4; ++ni)
#pragma unroll
      for (int v = 0; v < 8; ++v) {
        int row = m0 + waveM * 32 + mi * 16 + lh * 8 + v;
        int col = n0 + waveN * 64 + ni * 16 + lm;
        C[(size_t)row * N + col] = acc[mi][ni][v];
      }
}

// ---------------------------------------------------------------------------
// Conversion / padding helpers (weights -> bf16, pre-transposed to [N x K])
// ---------------------------------------------------------------------------
__global__ void cvt_f32_bf16(const float* __restrict__ in, __bf16* __restrict__ o, int n) {
  int i = blockIdx.x * blockDim.x + threadIdx.x;
  if (i < n) o[i] = (__bf16)in[i];
}

// in[K x N] f32 -> out[N x K] bf16
__global__ void cvt_transpose_bf16(const float* __restrict__ in, __bf16* __restrict__ o,
                                   int K, int N) {
  int i = blockIdx.x * blockDim.x + threadIdx.x;
  if (i >= N * K) return;
  int n = i / K, k = i - n * K;
  o[i] = (__bf16)in[(size_t)k * N + n];
}

// Wx [1536 x 80] -> bf16 transposed+padded [128 x 1536]
__global__ void pad_wx_t(const float* __restrict__ in, __bf16* __restrict__ o) {
  int i = blockIdx.x * blockDim.x + threadIdx.x;
  if (i >= N_XDBL_PAD * D_INNER) return;
  int n = i / D_INNER, k = i - n * D_INNER;
  o[i] = (__bf16)(n < (DT_RANK + 2 * D_STATE) ? in[(size_t)k * (DT_RANK + 2 * D_STATE) + n] : 0.f);
}

// Wdt [48 x 1536] -> bf16 transposed+padded [1536 x 64]
__global__ void pad_wdt_t(const float* __restrict__ in, __bf16* __restrict__ o) {
  int i = blockIdx.x * blockDim.x + threadIdx.x;
  if (i >= D_INNER * K_DT_PAD) return;
  int n = i >> 6, k = i & (K_DT_PAD - 1);
  o[i] = (__bf16)(k < DT_RANK ? in[(size_t)k * D_INNER + n] : 0.f);
}

// x_dbl [8192 x 128] cols 0..47 -> bf16 padded [8192 x 64] (A matrix, row-major)
__global__ void pad_dt(const float* __restrict__ xdbl, __bf16* __restrict__ o) {
  int i = blockIdx.x * blockDim.x + threadIdx.x;
  if (i >= M_ROWS * K_DT_PAD) return;
  int c = i & (K_DT_PAD - 1), r = i >> 6;
  o[i] = (__bf16)(c < DT_RANK ? xdbl[(size_t)r * N_XDBL_PAD + c] : 0.f);
}

// Depthwise causal conv(4) + bias + SiLU. Reads x-half of xz; writes f32 + bf16.
__global__ void conv_silu(const float* __restrict__ xz, const float* __restrict__ w,
                          const float* __restrict__ bias, float* __restrict__ xc,
                          __bf16* __restrict__ xcb) {
  int i = blockIdx.x * blockDim.x + threadIdx.x;
  if (i >= M_ROWS * D_INNER) return;
  int d = i % D_INNER, row = i / D_INNER, tpos = row & (L_SEQ - 1);
  float s = bias[d];
#pragma unroll
  for (int j = 0; j < D_CONV; ++j) {
    int tt = tpos + j - (D_CONV - 1);
    if (tt >= 0) s += w[d * D_CONV + j] * xz[(size_t)(row + j - (D_CONV - 1)) * N_XZ + d];
  }
  float v = s / (1.f + __expf(-s));    // SiLU
  xc[i]  = v;
  xcb[i] = (__bf16)v;
}

// ---------------------------------------------------------------------------
// Selective scan: one thread per channel d, 16 recurrent states in registers,
// sequential over L. B_t/C_t (32 floats) broadcast through LDS each step.
// Fuses softplus(delta_pre + bdt). Writes y_pre into the dead x-half of xz.
// ---------------------------------------------------------------------------
__global__ __launch_bounds__(256)
void selective_scan(const float* __restrict__ xc, const float* __restrict__ deltap,
                    const float* __restrict__ xdbl, const float* __restrict__ bdt,
                    const float* __restrict__ A_log, const float* __restrict__ Dp,
                    float* __restrict__ xz /* y_pre -> x-half */) {
  const int d  = blockIdx.x * blockDim.x + threadIdx.x;   // 0..1535
  const int bb = blockIdx.y;                               // batch

  float A[D_STATE], s[D_STATE];
#pragma unroll
  for (int n = 0; n < D_STATE; ++n) {
    A[n] = -__expf(A_log[d * D_STATE + n]);
    s[n] = 0.f;
  }
  const float bdt_d = bdt[d];
  const float Dd    = Dp[d];

  __shared__ float BC[2 * D_STATE];

  for (int tpos = 0; tpos < L_SEQ; ++tpos) {
    const int row = bb * L_SEQ + tpos;
    if (threadIdx.x < 2 * D_STATE)
      BC[threadIdx.x] = xdbl[(size_t)row * N_XDBL_PAD + DT_RANK + threadIdx.x];
    __syncthreads();

    float dp    = deltap[(size_t)row * D_INNER + d] + bdt_d;
    float delta = (dp > 20.f) ? dp : log1pf(__expf(dp));   // softplus
    float u     = xc[(size_t)row * D_INNER + d];
    float du    = delta * u;
    float y     = 0.f;
#pragma unroll
    for (int n = 0; n < D_STATE; ++n) {
      float dA = __expf(delta * A[n]);
      s[n] = dA * s[n] + du * BC[n];
      y   += s[n] * BC[D_STATE + n];
    }
    y += u * Dd;
    xz[(size_t)row * N_XZ + d] = y;   // y_pre into dead x-half
    __syncthreads();
  }
}

// y = y_pre * silu(z) -> bf16 (for out-proj GEMM)
__global__ void gate_silu(const float* __restrict__ xz, __bf16* __restrict__ yb) {
  int i = blockIdx.x * blockDim.x + threadIdx.x;
  if (i >= M_ROWS * D_INNER) return;
  int d = i % D_INNER, row = i / D_INNER;
  float yp = xz[(size_t)row * N_XZ + d];
  float z  = xz[(size_t)row * N_XZ + D_INNER + d];
  float g  = z / (1.f + __expf(-z));
  yb[i] = (__bf16)(yp * g);
}

// ---------------------------------------------------------------------------
// Launch
// ---------------------------------------------------------------------------
extern "C" void kernel_launch(void* const* d_in, const int* in_sizes, int n_in,
                              void* d_out, int out_size, void* d_ws, size_t ws_size,
                              hipStream_t stream) {
  const float* hidden = (const float*)d_in[0];
  const float* Win    = (const float*)d_in[1];
  const float* conv_w = (const float*)d_in[2];
  const float* conv_b = (const float*)d_in[3];
  const float* Wx     = (const float*)d_in[4];
  const float* Wdt    = (const float*)d_in[5];
  const float* bdt    = (const float*)d_in[6];
  const float* A_log  = (const float*)d_in[7];
  const float* Dp     = (const float*)d_in[8];
  const float* Wout   = (const float*)d_in[9];
  float* out = (float*)d_out;

  // workspace layout (256B aligned slabs)
  char* ws = (char*)d_ws;
  size_t off = 0;
  auto slab = [&](size_t bytes) {
    size_t o = off;
    off = (off + bytes + 255) & ~(size_t)255;
    return (void*)(ws + o);
  };
  __bf16* hiddenB = (__bf16*)slab((size_t)M_ROWS * D_MODEL * 2);
  __bf16* WinT    = (__bf16*)slab((size_t)N_XZ * D_MODEL * 2);        // [3072 x 768]
  __bf16* WxT     = (__bf16*)slab((size_t)N_XDBL_PAD * D_INNER * 2);  // [128 x 1536]
  __bf16* WdtT    = (__bf16*)slab((size_t)D_INNER * K_DT_PAD * 2);    // [1536 x 64]
  __bf16* WoutT   = (__bf16*)slab((size_t)D_MODEL * D_INNER * 2);     // [768 x 1536]
  float*  xz      = (float*)slab((size_t)M_ROWS * N_XZ * 4);
  float*  xconv   = (float*)slab((size_t)M_ROWS * D_INNER * 4);
  __bf16* xconvB  = (__bf16*)slab((size_t)M_ROWS * D_INNER * 2);      // reused as y_bf16
  float*  xdbl    = (float*)slab((size_t)M_ROWS * N_XDBL_PAD * 4);
  __bf16* dtB     = (__bf16*)slab((size_t)M_ROWS * K_DT_PAD * 2);
  float*  deltap  = (float*)slab((size_t)M_ROWS * D_INNER * 4);

  const int TB = 256;
  auto blocks = [](long n) { return (unsigned)((n + 255) / 256); };

  // 1) convert/transposes -> bf16
  cvt_f32_bf16<<<blocks((long)M_ROWS * D_MODEL), TB, 0, stream>>>(hidden, hiddenB, M_ROWS * D_MODEL);
  cvt_transpose_bf16<<<blocks((long)N_XZ * D_MODEL), TB, 0, stream>>>(Win, WinT, D_MODEL, N_XZ);
  pad_wx_t<<<blocks((long)N_XDBL_PAD * D_INNER), TB, 0, stream>>>(Wx, WxT);
  pad_wdt_t<<<blocks((long)D_INNER * K_DT_PAD), TB, 0, stream>>>(Wdt, WdtT);
  cvt_transpose_bf16<<<blocks((long)D_MODEL * D_INNER), TB, 0, stream>>>(Wout, WoutT, D_INNER, D_MODEL);

  // 2) GEMM1: xz = hidden @ Win    [8192x768]@[768x3072]
  gemm_bf16_wmma<<<dim3(N_XZ / 128, M_ROWS / 128), TB, 0, stream>>>(
      hiddenB, WinT, xz, M_ROWS, N_XZ, D_MODEL);

  // 3) depthwise conv + SiLU
  conv_silu<<<blocks((long)M_ROWS * D_INNER), TB, 0, stream>>>(xz, conv_w, conv_b, xconv, xconvB);

  // 4) GEMM2: x_dbl = xconv @ Wx_pad   [8192x1536]@[1536x128]
  gemm_bf16_wmma<<<dim3(N_XDBL_PAD / 128, M_ROWS / 128), TB, 0, stream>>>(
      xconvB, WxT, xdbl, M_ROWS, N_XDBL_PAD, D_INNER);

  // 5) dt slice -> bf16 padded K=64
  pad_dt<<<blocks((long)M_ROWS * K_DT_PAD), TB, 0, stream>>>(xdbl, dtB);

  // 6) GEMM3: delta_pre = dt @ Wdt_pad  [8192x64]@[64x1536]
  gemm_bf16_wmma<<<dim3(D_INNER / 128, M_ROWS / 128), TB, 0, stream>>>(
      dtB, WdtT, deltap, M_ROWS, D_INNER, K_DT_PAD);

  // 7) selective scan (softplus fused); y_pre overwrites x-half of xz
  selective_scan<<<dim3(D_INNER / 256, B_SZ), 256, 0, stream>>>(
      xconv, deltap, xdbl, bdt, A_log, Dp, xz);

  // 8) gating: y = y_pre * silu(z) -> bf16 (reuse xconvB slab)
  gate_silu<<<blocks((long)M_ROWS * D_INNER), TB, 0, stream>>>(xz, xconvB);

  // 9) GEMM4: out = y @ Wout   [8192x1536]@[1536x768]
  gemm_bf16_wmma<<<dim3(D_MODEL / 128, M_ROWS / 128), TB, 0, stream>>>(
      xconvB, WoutT, out, M_ROWS, D_MODEL, D_INNER);
}
